// VectorQuantizer_WeightCodebookLoss_16329465659951
// MI455X (gfx1250) — matrix operand order repaired and checked by reading, not compiled, not run on hardware
//
#include <hip/hip_runtime.h>

// ---------------------------------------------------------------------------
// VQ-VAE vector quantizer for MI455X (gfx1250, wave32, WMMA).
//   z:        (16, 512, 64, 64) f32   -> 65536 rows of dim 512 (strided!)
//   codebook: (1024, 512) f32
// Distance GEMM: v_wmma_f32_16x16x32_f16 (f16 in, f32 accumulate).
// B-tiles staged to LDS once per block via async global->LDS copies
// (ASYNCcnt), double-buffered, shared by all 8 waves -> 8x less L2 traffic.
// ---------------------------------------------------------------------------

typedef __attribute__((ext_vector_type(16))) _Float16 v16h;
typedef __attribute__((ext_vector_type(8)))  _Float16 v8h;
typedef __attribute__((ext_vector_type(8)))  float    v8f;

static constexpr int    N_E    = 1024;
static constexpr int    E_DIM  = 512;
static constexpr int    HW     = 64 * 64;          // 4096
static constexpr int    NROWS  = 16 * HW;          // 65536 flattened z vectors
static constexpr size_t ZQ_ELEMS = (size_t)16 * E_DIM * HW;   // 33554432
static constexpr size_t LOSS_OFF = ZQ_ELEMS;
static constexpr size_t IDX_OFF  = ZQ_ELEMS + 1;

// workspace layout (bytes)
static constexpr size_t WS_CB16   = 0;                          // 1024*512 f16 = 1 MB
static constexpr size_t WS_CNORM  = (size_t)N_E * E_DIM * 2;    // 1024 f32
static constexpr size_t WS_ARGMIN = WS_CNORM + (size_t)N_E * 4; // 65536 i32
static constexpr size_t WS_PART   = WS_ARGMIN + (size_t)NROWS * 4; // 1024 f32

// ---------------------------------------------------------------------------
// CDNA5 async global->LDS copy (16B per lane, tracked by ASYNCcnt).
// Builtin signature (leaked by clang diagnostic): param0 is
// '__attribute__((vector_size(16))) int __device__ *' (GCC vector, AS1),
// param1 is the LDS-side pointer (AS3). Pointers built via integer casts:
// flat==global for AS1; low 32 bits of a flat LDS address are the LDS offset.
// ---------------------------------------------------------------------------
typedef int v4i __attribute__((vector_size(16)));
typedef __attribute__((address_space(1))) v4i* as1_v4i_p;
typedef __attribute__((address_space(3))) v4i* as3_v4i_p;

__device__ __forceinline__ void async_copy16(const _Float16* g, _Float16* l) {
#if __has_builtin(__builtin_amdgcn_global_load_async_to_lds_b128)
  __builtin_amdgcn_global_load_async_to_lds_b128(
      (as1_v4i_p)(uintptr_t)g,
      (as3_v4i_p)(unsigned)(uintptr_t)l,
      0, 0);
#else
  asm volatile("global_load_async_to_lds_b128 %0, %1, off"
               :: "v"((unsigned)(uintptr_t)l),
                  "v"((unsigned long long)(uintptr_t)g)
               : "memory");
#endif
}

__device__ __forceinline__ void wait_async0() {
#if __has_builtin(__builtin_amdgcn_s_wait_asynccnt)
  __builtin_amdgcn_s_wait_asynccnt(0);
#else
  asm volatile("s_wait_asynccnt 0x0" ::: "memory");
#endif
}

// ---------------------------------------------------------------------------
// Kernel 1: codebook f32 -> f16, plus ||c_f16||^2 in f32 (self-consistent
// with the f16 WMMA dot products used for the argmin).
// ---------------------------------------------------------------------------
__global__ __launch_bounds__(256) void vq_prep(const float* __restrict__ cb,
                                               _Float16* __restrict__ cb16,
                                               float* __restrict__ cnorm) {
  __shared__ float red[256];
  const int row = blockIdx.x;
  const int t   = threadIdx.x;
  const float*    src = cb   + (size_t)row * E_DIM;
  _Float16*       dst = cb16 + (size_t)row * E_DIM;
  float s = 0.0f;
  for (int c = t; c < E_DIM; c += 256) {
    float v = src[c];
    _Float16 h = (_Float16)v;      // RTNE
    dst[c] = h;
    float hv = (float)h;
    s += hv * hv;
  }
  red[t] = s;
  __syncthreads();
  for (int o = 128; o > 0; o >>= 1) {
    if (t < o) red[t] += red[t + o];
    __syncthreads();
  }
  if (t == 0) cnorm[row] = red[0];
}

// ---------------------------------------------------------------------------
// Kernel 2: distance GEMM + argmin.
// Block = 256 threads (8 waves), M-tile = 128 rows, full N sweep (1024).
// Phase 1: z-tile -> LDS as f16 (padded 520-half stride = 4-bank row skew),
//          A fragments hoisted to VGPRs (16 x v16h per wave).
// Phase 2: A-staging LDS is dead -> reuse it as double-buffered 16-code
//          B-tiles (16x520 halves each), filled by async global->LDS copies.
// ---------------------------------------------------------------------------
static constexpr int MROWS = 128;
static constexpr int LDSW  = 520;                       // halves per row
static constexpr int BTILE_BYTES = 16 * LDSW * 2;       // 16,640 B per B-tile

__global__ __launch_bounds__(256) void vq_argmin_k(const float* __restrict__ z,
                                                   const _Float16* __restrict__ cb16,
                                                   const float* __restrict__ cnorm,
                                                   int* __restrict__ argmin) {
  __shared__ __align__(16) char smem[MROWS * LDSW * 2];   // 133,120 B
  _Float16* zt = (_Float16*)smem;                          // phase-1 alias

  const int t   = threadIdx.x;
  const int n0  = blockIdx.x * MROWS;     // first flattened z row of this block
  const int b   = n0 >> 12;               // / 4096  (128 | 4096 -> same batch)
  const int hw0 = n0 & (HW - 1);
  const float* zb = z + (size_t)b * E_DIM * HW + hw0;

  // phase-2 alias: double-buffered B-tiles (runtime address math only; a
  // constant-initialized pointer array would emit an addrspacecast in a
  // static initializer, which ld.lld rejects)
  auto bufptr = [&](int i) -> _Float16* {
    return (_Float16*)(smem + i * BTILE_BYTES);
  };

  // ---- phase 1: stage z tile to LDS as f16 (lanes coalesce over hw)
  {
    const int r  = t & (MROWS - 1);
    const int cp = t >> 7;                // 0 or 1
    _Float16* drow = &zt[r * LDSW];
    for (int j = 0; j < E_DIM / 2; ++j) {
      int c = cp + 2 * j;
      drow[c] = (_Float16)zb[(size_t)c * HW + r];
    }
  }
  __syncthreads();

  const int lane  = t & 31;
  const int wave  = t >> 5;
  const int lmod  = lane & 15;            // N column / M row selector
  const int lhalf = lane >> 4;            // K sub-block selector

  // ---- A fragments (wave-private 16 rows), ISA 16-bit A 16x32 layout:
  // halves 0..7 -> K = kb*32 + lhalf*8 + i, halves 8..15 -> +16.
  v16h afrag[16];
  {
    const _Float16* arow = &zt[(wave * 16 + lmod) * LDSW];
#pragma unroll
    for (int kb = 0; kb < 16; ++kb) {
      const _Float16* p = arow + kb * 32 + lhalf * 8;
      v8h lo = *(const v8h*)(p);          // 16B aligned (row stride 1040B)
      v8h hi = *(const v8h*)(p + 16);
      afrag[kb] = __builtin_shufflevector(lo, hi,
          0, 1, 2, 3, 4, 5, 6, 7, 8, 9, 10, 11, 12, 13, 14, 15);
    }
  }
  __syncthreads();                        // A region now dead -> B buffers

  float bestVal[8];
  int   bestIdx[8];
#pragma unroll
  for (int j = 0; j < 8; ++j) { bestVal[j] = 3.4e38f; bestIdx[j] = 0; }

  // ---- B-tile producer: 16 codes x 512 halves = 16 KB = 1024 x 16B chunks,
  //      4 chunks per thread, scattered into the padded LDS layout.
  auto issue_btile = [&](int nb, _Float16* buf) {
    const _Float16* src = cb16 + (size_t)nb * 16 * E_DIM;
#pragma unroll
    for (int i = 0; i < 4; ++i) {
      int k    = t + i * 256;
      int code = k >> 6;                  // 0..15
      int co   = (k & 63) * 8;            // half offset 0..504
      async_copy16(src + code * E_DIM + co, buf + code * LDSW + co);
    }
  };

  issue_btile(0, bufptr(0));
  wait_async0();
  __syncthreads();

  // ---- sweep all 1024 codes, 16 at a time, double-buffered
  for (int nb = 0; nb < N_E / 16; ++nb) {
    _Float16* cur = bufptr(nb & 1);
    if (nb + 1 < N_E / 16) issue_btile(nb + 1, bufptr((nb + 1) & 1));

    const int ncol = nb * 16 + lmod;      // this lane's candidate column
    const float cn = cnorm[ncol];
    // B 32x16 layout: halves 0..7 -> K = lhalf*8 + i, halves 8..15 -> +16 (N = code lmod)
    const _Float16* bp = cur + lmod * LDSW + lhalf * 8;

    v8f acc0 = {};
    v8f acc1 = {};
#pragma unroll
    for (int kb = 0; kb < 16; kb += 2) {
      {
        v8h lo = *(const v8h*)(bp + kb * 32);
        v8h hi = *(const v8h*)(bp + kb * 32 + 16);
        v16h bfrag = __builtin_shufflevector(lo, hi,
            0, 1, 2, 3, 4, 5, 6, 7, 8, 9, 10, 11, 12, 13, 14, 15);
        acc0 = __builtin_amdgcn_wmma_f32_16x16x32_f16(
            false, afrag[kb], false, bfrag, (short)0, acc0, false, false);
      }
      {
        v8h lo = *(const v8h*)(bp + (kb + 1) * 32);
        v8h hi = *(const v8h*)(bp + (kb + 1) * 32 + 16);
        v16h bfrag = __builtin_shufflevector(lo, hi,
            0, 1, 2, 3, 4, 5, 6, 7, 8, 9, 10, 11, 12, 13, 14, 15);
        acc1 = __builtin_amdgcn_wmma_f32_16x16x32_f16(
            false, afrag[kb + 1], false, bfrag, (short)0, acc1, false, false);
      }
    }

    // d = ||c||^2 - 2 <z, c>   (||z||^2 constant per row, drops from argmin)
#pragma unroll
    for (int j = 0; j < 8; ++j) {
      float d = cn - 2.0f * (acc0[j] + acc1[j]);
      if (d < bestVal[j]) { bestVal[j] = d; bestIdx[j] = ncol; }
    }

    wait_async0();                        // next tile landed in LDS
    __syncthreads();                      // publish; all readers of cur done
  }

  // ---- reduce across the 16 lanes that share each M row
  // C layout: VGPR j -> lanes 0-15 hold (M=j, N=lane), lanes 16-31 (M=j+8).
#pragma unroll
  for (int j = 0; j < 8; ++j) {
    float v  = bestVal[j];
    int   id = bestIdx[j];
#pragma unroll
    for (int m = 1; m <= 8; m <<= 1) {
      float ov = __shfl_xor(v, m, 32);
      int   oi = __shfl_xor(id, m, 32);
      if (ov < v || (ov == v && oi < id)) { v = ov; id = oi; }  // first-index ties
    }
    if (lmod == 0) {
      int M = lhalf ? (j + 8) : j;
      argmin[n0 + wave * 16 + M] = id;
    }
  }
}

// ---------------------------------------------------------------------------
// Kernel 3: gather z_q = codebook_f32[idx] into (B,C,H,W), indices output,
// and per-block partial sums of (z_q - z)^2 (deterministic, no fp atomics).
// Block handles 64 hw-consecutive rows -> lanes coalesce along hw.
// ---------------------------------------------------------------------------
__global__ __launch_bounds__(256) void vq_gather(const float* __restrict__ z,
                                                 const float* __restrict__ cb,
                                                 const int* __restrict__ argmin,
                                                 float* __restrict__ out,
                                                 float* __restrict__ partials) {
  __shared__ float red[256];
  const int t   = threadIdx.x;
  const int n0  = blockIdx.x * 64;
  const int b   = n0 >> 12;
  const int hw0 = n0 & (HW - 1);
  const int r   = t & 63;                 // row within block (hw-consecutive)
  const int c0  = t >> 6;                 // 4 c-slots
  const int n   = n0 + r;

  const int idx = argmin[n];
  const float* q  = cb + (size_t)idx * E_DIM;
  const float* zp = z   + (size_t)b * E_DIM * HW + hw0 + r;
  float*       op = out + (size_t)b * E_DIM * HW + hw0 + r;

  float s = 0.0f;
  for (int j = 0; j < E_DIM / 4; ++j) {
    int c = c0 + j * 4;
    float qv = q[c];                      // 1024 hot rows -> L2 resident
    float zv = zp[(size_t)c * HW];
    op[(size_t)c * HW] = qv;              // straight-through fwd == z_q
    float d = qv - zv;
    s += d * d;
  }
  if (c0 == 0) out[IDX_OFF + n] = (float)idx;

  red[t] = s;
  __syncthreads();
  for (int o = 128; o > 0; o >>= 1) {
    if (t < o) red[t] += red[t + o];
    __syncthreads();
  }
  if (t == 0) partials[blockIdx.x] = red[0];
}

// ---------------------------------------------------------------------------
// Kernel 4: deterministic final reduction -> codebook_loss = 1.25 * MSE
// ---------------------------------------------------------------------------
__global__ __launch_bounds__(256) void vq_loss(const float* __restrict__ partials,
                                               float* __restrict__ out) {
  __shared__ float red[256];
  const int t = threadIdx.x;
  float s = 0.0f;
#pragma unroll
  for (int j = 0; j < 4; ++j) s += partials[t + j * 256];   // fixed order
  red[t] = s;
  __syncthreads();
  for (int o = 128; o > 0; o >>= 1) {
    if (t < o) red[t] += red[t + o];
    __syncthreads();
  }
  if (t == 0) {
    // q_latent + BETA*e_latent == (1+0.25) * mean over 16*512*64*64 elements
    out[LOSS_OFF] = red[0] * 1.25f / 33554432.0f;
  }
}

// ---------------------------------------------------------------------------
extern "C" void kernel_launch(void* const* d_in, const int* in_sizes, int n_in,
                              void* d_out, int out_size, void* d_ws, size_t ws_size,
                              hipStream_t stream) {
  (void)in_sizes; (void)n_in; (void)out_size; (void)ws_size;
  const float* z  = (const float*)d_in[0];
  const float* cb = (const float*)d_in[1];
  float* out = (float*)d_out;

  char* ws = (char*)d_ws;
  _Float16* cb16    = (_Float16*)(ws + WS_CB16);
  float*    cnorm   = (float*)   (ws + WS_CNORM);
  int*      argmin  = (int*)     (ws + WS_ARGMIN);
  float*    partials= (float*)   (ws + WS_PART);

  vq_prep    <<<N_E,         256, 0, stream>>>(cb, cb16, cnorm);
  vq_argmin_k<<<NROWS/MROWS, 256, 0, stream>>>(z, cb16, cnorm, argmin);
  vq_gather  <<<NROWS/64,    256, 0, stream>>>(z, cb, argmin, out, partials);
  vq_loss    <<<1,           256, 0, stream>>>(partials, out);
}
